// EDCN_74878459838937
// MI455X (gfx1250) — matrix operand order repaired and checked by reading, not compiled, not run on hardware
//
#include <hip/hip_runtime.h>
#include <math.h>

// ---------------------------------------------------------------------------
// EDCN forward for MI455X (gfx1250, wave32, WMMA).
// Memory-bound op (~40 FLOP/B vs multi-hundred ridge) -> bf16 WMMA GEMMs with
// f32 accumulation, fused norm+GELU into GEMM A-staging, fused softmax into
// the sampling kernel, padded x_proj so bilinear gather needs no remapping.
// ---------------------------------------------------------------------------

typedef __bf16 bf16_t;
typedef __attribute__((ext_vector_type(16))) __bf16 v16bf;
typedef __attribute__((ext_vector_type(8)))  __bf16 v8bf;
typedef __attribute__((ext_vector_type(8)))  float  v8f;

#define CDIM   128
#define HDIM   80
#define WDIM   80
#define NSAMP  8
#define HPAD   82
#define WPAD   82
#define MROWS  (NSAMP * HDIM * WDIM)     // 51200 output rows for every GEMM
#define PIX    (HDIM * WDIM)             // 6400

static __device__ __forceinline__ bf16_t f2bf(float f) {
    // round-to-nearest-even f32 -> bf16
    unsigned int u = __builtin_bit_cast(unsigned int, f);
    unsigned int r = (u + 0x7FFFu + ((u >> 16) & 1u)) >> 16;
    return __builtin_bit_cast(bf16_t, (unsigned short)r);
}

// ---------------------------------------------------------------------------
// Generic WMMA GEMM:  Out = act(A[M x 128]) @ W[128 x Nc] + bias
//   NT         : number of 16-wide N tiles (Nc padded to NT*16 with zeros)
//   STORE_PAD  : 0 -> row-major Out[m*Nc + n];  1 -> padded NHWC (N,82,82,128)
//   DO_NORMGELU: fold per-sample layernorm + exact GELU into A staging
// Block = 256 threads = 8 waves; block tile = 128 rows x NT*16 cols.
// ---------------------------------------------------------------------------
template <int NT, int STORE_PAD, int DO_NORMGELU>
__global__ __launch_bounds__(256)
void k_wmma_gemm(const float* __restrict__ A, const float* __restrict__ Wm,
                 const float* __restrict__ bias, float* __restrict__ Out,
                 int Nc,
                 const float* __restrict__ stats,   // [2*NSAMP] mean, rstd
                 const float* __restrict__ gamma,
                 const float* __restrict__ beta)
{
    __shared__ __align__(16) bf16_t As[128][CDIM];      // 32 KB
    __shared__ __align__(16) bf16_t Wt[NT * 16][CDIM];  // transposed W, <=32 KB

    const int tid    = threadIdx.x;
    const int lane   = tid & 31;
    const int wave   = tid >> 5;
    const int blockM = blockIdx.x * 128;
    const int NcPad  = NT * 16;

    // ---- stage A tile (128 x 128 f32 -> bf16), optional norm+GELU ----
    for (int i = tid; i < 128 * CDIM; i += 256) {
        const int r = i >> 7;
        const int k = i & 127;
        const int m = blockM + r;
        float v = A[(size_t)m * CDIM + k];
        if (DO_NORMGELU) {
            const int s = m / PIX;
            v = (v - stats[2 * s]) * stats[2 * s + 1] * gamma[k] + beta[k];
            v = 0.5f * v * (1.0f + erff(v * 0.70710678118654752f)); // exact GELU
        }
        As[r][k] = f2bf(v);
    }
    // ---- stage W transposed: Wt[n][k] (K contiguous per output column) ----
    for (int i = tid; i < NcPad * CDIM; i += 256) {
        const int k = i / NcPad;
        const int n = i - k * NcPad;
        const float v = (n < Nc) ? Wm[k * Nc + n] : 0.0f;
        Wt[n][k] = f2bf(v);
    }
    __syncthreads();

    v8f acc[NT] = {};

    const int mrow  = wave * 16 + (lane & 15);
    const int ahalf = (lane >> 4) * 8;    // A: lanes 0-15 hold K 0..7/16..23, lanes 16-31 hold K 8..15/24..31
    const int bhalf = (lane >> 4) * 8;    // B: lanes 0-15 hold K 0..7/16..23, lanes 16-31 hold K 8..15/24..31
    const int ncol  = lane & 15;

#pragma unroll
    for (int kk = 0; kk < CDIM; kk += 32) {
        const v8bf a0 = *(const v8bf*)&As[mrow][kk + ahalf];
        const v8bf a1 = *(const v8bf*)&As[mrow][kk + 16 + ahalf];
        const v16bf af = __builtin_shufflevector(a0, a1,
            0,1,2,3,4,5,6,7,8,9,10,11,12,13,14,15);
#pragma unroll
        for (int t = 0; t < NT; ++t) {
            const v8bf b0 = *(const v8bf*)&Wt[t * 16 + ncol][kk + bhalf];
            const v8bf b1 = *(const v8bf*)&Wt[t * 16 + ncol][kk + 16 + bhalf];
            const v16bf bfv = __builtin_shufflevector(b0, b1,
                0,1,2,3,4,5,6,7,8,9,10,11,12,13,14,15);
            acc[t] = __builtin_amdgcn_wmma_f32_16x16x32_bf16(
                false, af, false, bfv, (short)0, acc[t], false, false);
        }
    }

    // ---- epilogue: bias add + store (C layout: VGPR r -> M = r + 8*(lane>=16)) ----
#pragma unroll
    for (int t = 0; t < NT; ++t) {
        const int nidx = t * 16 + ncol;
        if (nidx >= Nc) continue;
        const float bv = bias[nidx];
#pragma unroll
        for (int r = 0; r < 8; ++r) {
            const int m = blockM + wave * 16 + (lane >> 4) * 8 + r;
            const float v = acc[t][r] + bv;
            if (STORE_PAD) {
                const int s   = m / PIX;
                const int rem = m - s * PIX;
                const int h   = rem / WDIM;
                const int w   = rem - h * WDIM;
                Out[(((size_t)s * HPAD + (h + 1)) * WPAD + (w + 1)) * CDIM + nidx] = v;
            } else {
                Out[(size_t)m * Nc + nidx] = v;
            }
        }
    }
}

// ---------------------------------------------------------------------------
// Depthwise 3x3 conv ('same', zero pad) + bias -> x1, plus per-sample
// sum / sum-of-squares partials (for GroupNorm(1,C)) via block reduce+atomics.
// One thread per (n,h,w,c); a 256-thread block stays inside one sample.
// ---------------------------------------------------------------------------
__global__ __launch_bounds__(256)
void k_dwconv(const float* __restrict__ x, const float* __restrict__ dwk,
              const float* __restrict__ dwb, float* __restrict__ x1,
              float* __restrict__ sums)
{
    const int idx = blockIdx.x * 256 + threadIdx.x;
    const int c = idx & (CDIM - 1);
    int pix = idx >> 7;
    const int w = pix % WDIM; pix /= WDIM;
    const int h = pix % HDIM; pix /= HDIM;
    const int n = pix;

    float acc = dwb[c];
#pragma unroll
    for (int dy = 0; dy < 3; ++dy) {
        const int yy = h + dy - 1;
        if (yy < 0 || yy >= HDIM) continue;
#pragma unroll
        for (int dx = 0; dx < 3; ++dx) {
            const int xx = w + dx - 1;
            if (xx < 0 || xx >= WDIM) continue;
            acc += x[(((size_t)n * HDIM + yy) * WDIM + xx) * CDIM + c]
                 * dwk[(dy * 3 + dx) * CDIM + c];
        }
    }
    x1[idx] = acc;

    __shared__ float ssum[256];
    __shared__ float ssq[256];
    ssum[threadIdx.x] = acc;
    ssq[threadIdx.x]  = acc * acc;
    __syncthreads();
    for (int s = 128; s > 0; s >>= 1) {
        if (threadIdx.x < s) {
            ssum[threadIdx.x] += ssum[threadIdx.x + s];
            ssq[threadIdx.x]  += ssq[threadIdx.x + s];
        }
        __syncthreads();
    }
    if (threadIdx.x == 0) {
        atomicAdd(&sums[2 * n],     ssum[0]);
        atomicAdd(&sums[2 * n + 1], ssq[0]);
    }
}

__global__ void k_stats(const float* __restrict__ sums, float* __restrict__ stats)
{
    const int s = threadIdx.x;
    if (s < NSAMP) {
        const float cnt = (float)(PIX * CDIM);
        const float mu  = sums[2 * s] / cnt;
        const float var = sums[2 * s + 1] / cnt - mu * mu;
        stats[2 * s]     = mu;
        stats[2 * s + 1] = rsqrtf(var + 1e-6f);
    }
}

// ---------------------------------------------------------------------------
// Deformable sampling: one wave per (pixel, group), lane = channel-in-group.
// Softmax over the 9 taps in registers; 4 coalesced corner gathers per tap
// from the padded x_proj image; mask-weighted accumulate -> val[m][g*32+c].
// ---------------------------------------------------------------------------
__global__ __launch_bounds__(256)
void k_sample(const float* __restrict__ xp, const float* __restrict__ offs,
              const float* __restrict__ maskl, float* __restrict__ val)
{
    const int gid  = blockIdx.x * 8 + (threadIdx.x >> 5);   // over MROWS*4
    const int lane = threadIdx.x & 31;
    const int g = gid & 3;
    const int m = gid >> 2;
    int pix = m;
    const int w = pix % WDIM; pix /= WDIM;
    const int h = pix % HDIM; pix /= HDIM;
    const int n = pix;

    // softmax over P=9 tap logits
    float lg[9];
    float mx = -1e30f;
#pragma unroll
    for (int p = 0; p < 9; ++p) {
        lg[p] = maskl[(size_t)m * 36 + g * 9 + p];
        mx = fmaxf(mx, lg[p]);
    }
    float den = 0.0f;
#pragma unroll
    for (int p = 0; p < 9; ++p) { lg[p] = __expf(lg[p] - mx); den += lg[p]; }
    const float inv = 1.0f / den;

    // reference points normalized to the PADDED size (82 x 82)
    const float refx = ((w + 0.5f) / (float)WPAD) * 2.0f - 1.0f;
    const float refy = ((h + 0.5f) / (float)HPAD) * 2.0f - 1.0f;
    const float* img = xp + (size_t)n * HPAD * WPAD * CDIM + g * 32 + lane;

    float acc = 0.0f;
#pragma unroll
    for (int p = 0; p < 9; ++p) {
        const float ox = offs[(size_t)m * 72 + (g * 9 + p) * 2 + 0];
        const float oy = offs[(size_t)m * 72 + (g * 9 + p) * 2 + 1];
        const float ix = (ox + refx + 1.0f) * (0.5f * WPAD) - 0.5f;
        const float iy = (oy + refy + 1.0f) * (0.5f * HPAD) - 0.5f;
        const float x0f = floorf(ix), y0f = floorf(iy);
        const float wx1 = ix - x0f, wy1 = iy - y0f;
        const float wx0 = 1.0f - wx1, wy0 = 1.0f - wy1;
        const int x0 = (int)x0f, y0 = (int)y0f;

        auto corner = [&](int xi, int yi, float wgt) -> float {
            if (xi < 0 || xi > WPAD - 1 || yi < 0 || yi > HPAD - 1) return 0.0f;
            return img[((size_t)yi * WPAD + xi) * CDIM] * wgt;
        };
        const float sv = corner(x0,     y0,     wx0 * wy0)
                       + corner(x0 + 1, y0,     wx1 * wy0)
                       + corner(x0,     y0 + 1, wx0 * wy1)
                       + corner(x0 + 1, y0 + 1, wx1 * wy1);
        acc += (lg[p] * inv) * sv;
    }
    val[(size_t)m * CDIM + g * 32 + lane] = acc;
}

// ---------------------------------------------------------------------------
extern "C" void kernel_launch(void* const* d_in, const int* in_sizes, int n_in,
                              void* d_out, int out_size, void* d_ws, size_t ws_size,
                              hipStream_t stream)
{
    const float* x      = (const float*)d_in[0];
    const float* w_in   = (const float*)d_in[1];
    const float* b_in   = (const float*)d_in[2];
    const float* dwk    = (const float*)d_in[3];
    const float* dwb    = (const float*)d_in[4];
    const float* gamma  = (const float*)d_in[5];
    const float* beta   = (const float*)d_in[6];
    const float* w_off  = (const float*)d_in[7];
    const float* b_off  = (const float*)d_in[8];
    const float* w_mask = (const float*)d_in[9];
    const float* b_mask = (const float*)d_in[10];
    const float* w_out  = (const float*)d_in[11];
    const float* b_out  = (const float*)d_in[12];

    // workspace layout (floats); val aliases x1 (x1 dead after mask GEMM)
    float* ws    = (float*)d_ws;
    float* xp    = ws;                                 // N*82*82*128 = 6,885,376
    float* x1    = xp    + (size_t)NSAMP * HPAD * WPAD * CDIM;
    float* val   = x1;                                 // alias: reused after GEMMs
    float* offs  = x1    + (size_t)MROWS * CDIM;       // M*72
    float* maskl = offs  + (size_t)MROWS * 72;         // M*36
    float* sums  = maskl + (size_t)MROWS * 36;         // 16
    float* stats = sums  + 16;                         // 16

    hipMemsetAsync(xp,   0, (size_t)NSAMP * HPAD * WPAD * CDIM * sizeof(float), stream);
    hipMemsetAsync(sums, 0, 16 * sizeof(float), stream);

    const int gemm_blocks = MROWS / 128;               // 400

    // 1. input projection -> padded x_proj (bf16 WMMA)
    k_wmma_gemm<8, 1, 0><<<gemm_blocks, 256, 0, stream>>>(
        x, w_in, b_in, xp, 128, nullptr, nullptr, nullptr);

    // 2. depthwise conv + per-sample moment partials
    k_dwconv<<<(MROWS * CDIM) / 256, 256, 0, stream>>>(x, dwk, dwb, x1, sums);
    k_stats<<<1, 32, 0, stream>>>(sums, stats);

    // 3. offset projection (norm+GELU fused into A staging), Nc=72 -> 5 tiles
    k_wmma_gemm<5, 0, 1><<<gemm_blocks, 256, 0, stream>>>(
        x1, w_off, b_off, offs, 72, stats, gamma, beta);

    // 4. mask projection, Nc=36 -> 3 tiles
    k_wmma_gemm<3, 0, 1><<<gemm_blocks, 256, 0, stream>>>(
        x1, w_mask, b_mask, maskl, 36, stats, gamma, beta);

    // 5. deformable bilinear sampling + softmax-weighted sum (writes val = x1 space)
    k_sample<<<(MROWS * 4) / 8, 256, 0, stream>>>(xp, offs, maskl, val);

    // 6. output projection -> d_out
    k_wmma_gemm<8, 0, 0><<<gemm_blocks, 256, 0, stream>>>(
        val, w_out, b_out, (float*)d_out, 128, nullptr, nullptr, nullptr);
}
